// FeatureGenPytorch_91122026151937
// MI455X (gfx1250) — compile-verified, matrix operand order checked by reading, not run on hardware
//
#include <hip/hip_runtime.h>

typedef int v8i __attribute__((ext_vector_type(8)));

// ---------------------------------------------------------------------------
// Kernel 0: zero the two counters in workspace (graph-capture safe re-init)
// ---------------------------------------------------------------------------
__global__ void fg_init_kernel(int* __restrict__ counts) {
    if (threadIdx.x < 2) counts[threadIdx.x] = 0;
}

// ---------------------------------------------------------------------------
// Kernel 1: count non-zero (NaN treated as zero) entries of the left-hand and
// right-hand slices via V_WMMA_I32_16X16X64_IU8 (indicator bytes x ones).
//
// Flat element space: 32768 frames * 63 floats = 2,064,384 = 2016 chunks of
// 1024 (32 lanes * 32 bytes). 42 blocks * 256 threads = 336 waves, 6 chunks
// per wave exactly -> no tail, EXEC all-ones at every WMMA.
//
// Since B is all-ones, sum over all D elements = 16 * (count of 1-bytes in A),
// independent of the A VGPR layout, so we can pack bytes in load order.
// ---------------------------------------------------------------------------
__global__ __launch_bounds__(256) void fg_count_kernel(
    const float* __restrict__ x, int* __restrict__ counts) {

    const unsigned lane   = threadIdx.x & 31u;
    const unsigned wave   = (blockIdx.x * blockDim.x + threadIdx.x) >> 5;
    const unsigned nwaves = (gridDim.x * blockDim.x) >> 5;

    v8i accL = {};
    v8i accR = {};
    v8i ones;
#pragma unroll
    for (int i = 0; i < 8; ++i) ones[i] = 0x01010101;

    for (unsigned chunk = wave; chunk < 2016u; chunk += nwaves) {
        const unsigned base = chunk * 1024u;

        unsigned pL[8] = {0u, 0u, 0u, 0u, 0u, 0u, 0u, 0u};
        unsigned pR[8] = {0u, 0u, 0u, 0u, 0u, 0u, 0u, 0u};

#pragma unroll
        for (int j = 0; j < 32; ++j) {
            const unsigned idx   = base + (unsigned)j * 32u + lane; // flat in [0, 2064384)
            const unsigned frame = idx / 63u;                      // const-div -> mulhi
            const unsigned r     = idx - frame * 63u;
            const unsigned fbase = frame * 345u;

            const float vL = x[fbase + 120u + r];  // lefth  = x[:,40:61]
            const float vR = x[fbase + 282u + r];  // righth = x[:,94:115]

            // ordered compares are false for NaN -> NaN counted as zero
            const unsigned iL = (vL < 0.0f || vL > 0.0f) ? 1u : 0u;
            const unsigned iR = (vR < 0.0f || vR > 0.0f) ? 1u : 0u;

            pL[j >> 2] |= iL << ((j & 3) * 8);
            pR[j >> 2] |= iR << ((j & 3) * 8);
        }

        v8i aL, aR;
#pragma unroll
        for (int i = 0; i < 8; ++i) { aL[i] = (int)pL[i]; aR[i] = (int)pR[i]; }

        // D = A x Ones + C   (unsigned 8-bit operands, i32 accumulate: exact)
        accL = __builtin_amdgcn_wmma_i32_16x16x64_iu8(false, aL, false, ones,
                                                      accL, false, false);
        accR = __builtin_amdgcn_wmma_i32_16x16x64_iu8(false, aR, false, ones,
                                                      accR, false, false);
    }

    int sL = 0, sR = 0;
#pragma unroll
    for (int i = 0; i < 8; ++i) { sL += accL[i]; sR += accR[i]; }

    // wave32 cross-lane tree reduction
    for (int off = 16; off > 0; off >>= 1) {
        sL += __shfl_xor(sL, off, 32);
        sR += __shfl_xor(sR, off, 32);
    }

    if (lane == 0) {
        atomicAdd(&counts[0], sL >> 4);  // every A byte was counted 16x
        atomicAdd(&counts[1], sR >> 4);
    }
}

// ---------------------------------------------------------------------------
// Kernel 2: one block per output row t (0..99). Builds feat[t], feat[t+1] in
// LDS, then writes the 1196-float feature row.
// ---------------------------------------------------------------------------
__global__ __launch_bounds__(128) void fg_feat_kernel(
    const float* __restrict__ x, const int* __restrict__ counts,
    float* __restrict__ out) {

    __shared__ float f0[258];  // feat[t]   : 86 landmarks * 3
    __shared__ float f1[258];  // feat[t+1]

    const int  t        = blockIdx.x;
    const bool cond     = counts[0] > counts[1];
    const int  handBase = cond ? 40 : 94;

    // assemble feat rows t and t+1
    for (int i = threadIdx.x; i < 516; i += blockDim.x) {
        const int  fr = (i < 258) ? t : (t + 1);
        const int  k  = (i < 258) ? i : (i - 258);
        const int  lm = k / 3;
        const int  c  = k - lm * 3;

        int srcLm;
        if (lm < 21)      srcLm = handBase + lm;        // hand
        else if (lm < 46) srcLm = 61 + (lm - 21);       // pose
        else              srcLm = lm - 46;              // lip

        float v = x[fr * 345 + srcLm * 3 + c];
        if (v != v) v = 0.0f;                           // NaN -> 0
        if (c == 0 && cond) v = -v;                     // mirror x for left hand

        if (i < 258) f0[k] = v; else f1[k] = v;
    }
    __syncthreads();

    float* o = out + t * 1196;

    // feat slices [0,153) and dxyz slices [153,306)
    for (int k = threadIdx.x; k < 306; k += blockDim.x) {
        const int kk = (k < 153) ? k : (k - 153);
        int lm, c;
        if (kk < 63)       { lm = kk / 3;              c = kk - lm * 3; }
        else if (kk < 113) { int p = kk - 63;  lm = 21 + p / 2; c = p & 1; }
        else               { int p = kk - 113; lm = 46 + p / 2; c = p & 1; }
        float v = f0[lm * 3 + c];
        if (k >= 153) v -= f1[lm * 3 + c];              // dxyz = feat[t] - feat[t+1]
        o[k] = v;
    }

    // pairwise distances: hdist(210,3D) pdist(300,2D) oldist(190,2D) ildist(190,2D)
    for (int k = threadIdx.x; k < 890; k += blockDim.x) {
        int n, base, rem, use3d;
        if (k < 210)      { n = 21; base = 0;  rem = k;       use3d = 1; }
        else if (k < 510) { n = 25; base = 21; rem = k - 210; use3d = 0; }
        else if (k < 700) { n = 20; base = 46; rem = k - 510; use3d = 0; }
        else              { n = 20; base = 66; rem = k - 700; use3d = 0; }

        // unrank triu_indices(n, 1) row-major
        int i = 0, cnt = n - 1;
        while (rem >= cnt) { rem -= cnt; --cnt; ++i; }
        const int j = i + 1 + rem;

        const float* pi = &f0[(base + i) * 3];
        const float* pj = &f0[(base + j) * 3];
        const float dx = pi[0] - pj[0];
        const float dy = pi[1] - pj[1];
        float s = dx * dx + dy * dy;
        if (use3d) { const float dz = pi[2] - pj[2]; s += dz * dz; }

        o[306 + k] = sqrtf(s);
    }
}

// ---------------------------------------------------------------------------
extern "C" void kernel_launch(void* const* d_in, const int* in_sizes, int n_in,
                              void* d_out, int out_size, void* d_ws, size_t ws_size,
                              hipStream_t stream) {
    (void)in_sizes; (void)n_in; (void)out_size; (void)ws_size;

    const float* x      = (const float*)d_in[0];
    int*         counts = (int*)d_ws;
    float*       out    = (float*)d_out;

    fg_init_kernel<<<1, 32, 0, stream>>>(counts);
    fg_count_kernel<<<42, 256, 0, stream>>>(x, counts);  // 336 waves * 6 chunks = 2016
    fg_feat_kernel<<<100, 128, 0, stream>>>(x, counts, out);
}
